// SOM_37821482009424
// MI455X (gfx1250) — matrix operand order repaired and checked by reading, not compiled, not run on hardware
//
#include <hip/hip_runtime.h>

// SOM BMU: per (b,t) argmin_k ||w_k||^2 - 2 * x(b,t).w_k, one-hot output.
// cross term = batched GEMM via v_wmma_f32_16x16x32_bf16 (fp32 accumulate).
// Memory-bound: ~206 MB min traffic @ 23.3 TB/s ~= 8.8 us; bf16 WMMA compute ~4.6 us.

#define CDIM 224
#define TDIM 224
#define NBATCH 512
#define CPAD 228            // LDS row pitch in bf16 (456B = 114 dwords -> conflict-free)
#define ROWS_PER_PASS 112   // stage half the t-rows per pass: 112*228*2 = 51072 B LDS

typedef __attribute__((ext_vector_type(16))) __bf16 v16bf;
typedef __attribute__((ext_vector_type(8)))  float  v8f;

union FragA { uint4 u[2]; v16bf v; };   // 32B: two 16B chunks per lane
union FragB { uint2 q[4]; v16bf v; };   // 32B: four 8B chunks per lane (8B-aligned LDS)

__device__ __forceinline__ unsigned short f2bf(float f) {
    unsigned u = __float_as_uint(f);
    u += 0x7FFFu + ((u >> 16) & 1u);    // round-to-nearest-even
    return (unsigned short)(u >> 16);
}

// Prep: w (fp32, 224x224) -> bf16 copy + wsq[k] = ||w_k||^2
__global__ void som_prep(const float* __restrict__ w,
                         unsigned short* __restrict__ wbf,
                         float* __restrict__ wsq) {
    const int tid = threadIdx.x;
    for (int i = tid; i < CDIM * CDIM; i += 256) wbf[i] = f2bf(w[i]);
    for (int k = tid; k < CDIM; k += 256) {
        float s = 0.0f;
        const float* row = w + k * CDIM;
        for (int c = 0; c < CDIM; ++c) { float v = row[c]; s += v * v; }
        wsq[k] = s;
    }
}

__global__ __launch_bounds__(224, 2) void som_main(const float* __restrict__ inp,
                                                   const unsigned short* __restrict__ wbf,
                                                   const float* __restrict__ wsqg,
                                                   float* __restrict__ out) {
    __shared__ unsigned short ldsX[ROWS_PER_PASS * CPAD]; // xT[t_local][c] in bf16
    __shared__ float wsqs[CDIM];

    const int tid    = threadIdx.x;
    const int b      = blockIdx.x;
    const int lane   = tid & 31;
    const int wave   = tid >> 5;          // 0..6
    const int lanelo = lane & 15;
    const bool hi    = lane >= 16;

    const float* inb = inp + (size_t)b * (CDIM * TDIM);
    float* outb      = out + (size_t)b * (CDIM * TDIM);

    // Zero the whole output plane for this batch (streaming stores).
    {
        float4 z = make_float4(0.f, 0.f, 0.f, 0.f);
        float4* o4 = (float4*)outb;
        for (int i = tid; i < (CDIM * TDIM) / 4; i += 224) o4[i] = z;  // 56 iters
    }
    if (tid < CDIM) wsqs[tid] = wsqg[tid];
    __threadfence_block();  // zeros visible before later one-hot scatters

    #pragma unroll 1
    for (int pass = 0; pass < 2; ++pass) {
        __syncthreads();  // also covers zero-phase before first scatter
        // Stage + transpose + bf16-convert t-rows [pass*112, pass*112+112).
        for (int i = tid; i < CDIM * ROWS_PER_PASS; i += 224) {  // 112 iters
            int cc = i / ROWS_PER_PASS;
            int tt = i - cc * ROWS_PER_PASS;
            float v = inb[cc * TDIM + pass * ROWS_PER_PASS + tt];
            ldsX[tt * CPAD + cc] = f2bf(v);  // conflict-free: 114-dword row pitch
        }
        __syncthreads();

        // Each wave owns one 16-wide t-tile this pass.
        const int ttile = pass * 7 + wave;
        const int trow  = wave * 16 + lanelo;   // local LDS row for this lane's column n

        // B fragments (x tile, 32x16 K x N): lanes 0-15 hold K=0..15 (32B),
        // lanes 16-31 hold K=16..31. Load all 7 c-tiles once; keep register-resident.
        FragB bfr[7];
        {
            const unsigned short* xrow = &ldsX[trow * CPAD + (hi ? 16 : 0)];
            #pragma unroll
            for (int ct = 0; ct < 7; ++ct) {
                const unsigned short* p = xrow + ct * 32;
                bfr[ct].q[0] = *(const uint2*)(p);
                bfr[ct].q[1] = *(const uint2*)(p + 4);
                bfr[ct].q[2] = *(const uint2*)(p + 8);
                bfr[ct].q[3] = *(const uint2*)(p + 12);
            }
        }

        float bestd = 3.4e38f;
        int   bestk = 0;
        // A fragment (w tile, 16x32 M x K): lane row M = lanelo;
        // lanes 0-15: K in {0..15}; lanes 16-31: K in {16..31}.
        const unsigned short* wrow0 = wbf + lanelo * CDIM + (hi ? 16 : 0);

        // unroll 2: pipelines A-loads ~2 k-tiles deep without blowing VGPRs.
        #pragma unroll 2
        for (int kt = 0; kt < 14; ++kt) {
            v8f acc = {0.f, 0.f, 0.f, 0.f, 0.f, 0.f, 0.f, 0.f};
            const unsigned short* wr = wrow0 + kt * 16 * CDIM;
            #pragma unroll
            for (int ct = 0; ct < 7; ++ct) {
                FragA a;
                const unsigned short* p = wr + ct * 32;
                a.u[0] = *(const uint4*)(p);
                a.u[1] = *(const uint4*)(p + 8);
                acc = __builtin_amdgcn_wmma_f32_16x16x32_bf16(
                        false, a.v, false, bfr[ct].v, (short)0, acc, false, false);
            }
            // D layout: lanes 0-15 VGPR r -> M=r; lanes 16-31 -> M=8+r; N = lane%16.
            const int kb = kt * 16 + (hi ? 8 : 0);
            #pragma unroll
            for (int r = 0; r < 8; ++r) {
                float d = wsqs[kb + r] - 2.0f * acc[r];
                if (d < bestd) { bestd = d; bestk = kb + r; }
            }
        }
        // Merge the two k-halves living in lanes L and L^16 (same t column).
        float od  = __shfl_xor(bestd, 16, 32);
        int   okk = __shfl_xor(bestk, 16, 32);
        if (od < bestd || (od == bestd && okk < bestk)) { bestd = od; bestk = okk; }

        if (!hi) {
            int t = ttile * 16 + lanelo;
            outb[(size_t)bestk * TDIM + t] = 1.0f;  // one-hot scatter
        }
    }
}

extern "C" void kernel_launch(void* const* d_in, const int* in_sizes, int n_in,
                              void* d_out, int out_size, void* d_ws, size_t ws_size,
                              hipStream_t stream) {
    const float* inp = (const float*)d_in[0];   // (512, 224, 224) fp32
    const float* w   = (const float*)d_in[1];   // (224, 224) fp32
    float* out       = (float*)d_out;           // (512, 1, 224, 224) fp32

    // Workspace: [0, 100352) bf16 copy of w; [100352, 101248) wsq fp32.
    unsigned short* wbf = (unsigned short*)d_ws;
    float* wsq          = (float*)((char*)d_ws + CDIM * CDIM * 2);

    som_prep<<<1, 256, 0, stream>>>(w, wbf, wsq);
    som_main<<<NBATCH, 224, 0, stream>>>(inp, wbf, wsq, out);
}
